// VectorQuantizer_15101105012791
// MI455X (gfx1250) — matrix operand order repaired and checked by reading, not compiled, not run on hardware
//
#include <hip/hip_runtime.h>
#include <hip/hip_bf16.h>

typedef __attribute__((ext_vector_type(16))) __bf16 v16bf;
typedef __attribute__((ext_vector_type(8)))  float  v8f;
typedef __attribute__((ext_vector_type(4)))  unsigned int v4u;
typedef __attribute__((ext_vector_type(8)))  int v8i;
typedef __attribute__((ext_vector_type(4)))  int v4i;

#define N_TOKENS   131072
#define DIM        64
#define K_CODES    1024
#define N_KTILES   64            // 1024 / 16 code column tiles
#define CB_PACKED_BYTES 262144   // 64 t * 2 c * 2 s * 2 h * 32 lanes * 16 B
#define CB_PACKED_U4    16384    // /16
#define LDS_TOTAL  266240        // packed cb (256KB) + norms (4KB), contiguous
#define TDM_ELEMS  33280         // 266240 / 8-byte elements
// workspace layout (bytes) -- packed cb and norms are contiguous for one TDM op
#define WS_CB_PACK 0
#define WS_CB_NORM 262144
#define WS_HIST    266240
#define WS_PART    270336
#define NBLK_K2    512

// ---------------------------------------------------------------------------
// k_pack: split codebook f32 -> bf16 hi + bf16 lo, pre-swizzled into the
// WMMA B-matrix (32x16, 16-bit) lane layout, in contiguous k order (same as A):
//   element e of lane's v16bf:  k = 32*c + 16*(lane>=16) + e,  n = lane&15
// Stored as [t][c][s][h][lane][j] bf16 so k_argmin reads raw ds_load_b128.
// ---------------------------------------------------------------------------
__global__ __launch_bounds__(256) void k_pack(const float* __restrict__ cb,
                                              __bf16* __restrict__ out) {
  int id   = blockIdx.x * 256 + threadIdx.x;   // 0..131071
  int j    = id & 7;
  int lane = (id >> 3) & 31;
  int h    = (id >> 8) & 1;
  int s    = (id >> 9) & 1;
  int c    = (id >> 10) & 1;
  int t    = id >> 11;
  int e    = h * 8 + j;
  int k    = c * 32 + ((lane >> 4) << 4) + e;
  int code = t * 16 + (lane & 15);
  float v  = cb[code * DIM + k];
  __bf16 hi = (__bf16)v;
  out[id] = s ? (__bf16)(v - (float)hi) : hi;
}

// ---------------------------------------------------------------------------
// k_norm: |e_k|^2, zero histogram + partials (ws is not re-zeroed by harness)
// ---------------------------------------------------------------------------
__global__ __launch_bounds__(256) void k_norm(const float* __restrict__ cb,
                                              float* __restrict__ norm,
                                              int* __restrict__ hist,
                                              float* __restrict__ part) {
  int i = blockIdx.x * 256 + threadIdx.x;
  if (i < K_CODES) {
    const float* r = cb + i * DIM;
    float s = 0.f;
    #pragma unroll
    for (int d = 0; d < DIM; ++d) s += r[d] * r[d];
    norm[i] = s;
    hist[i] = 0;
  }
  if (i < NBLK_K2) part[i] = 0.f;
}

// ---------------------------------------------------------------------------
// k_argmin: full packed codebook + norms staged into 260KB LDS with a single
// Tensor Data Mover op (issued by wave 0, S_WAIT_TENSORCNT, then barrier).
// 256 threads = 8 waves; each wave processes 4 row-tiles of 16 tokens.
// x.e computed as hi*hi + hi*lo + lo*hi (3 bf16 WMMAs per K-chunk of 32).
// Inner sweep is software-pipelined: min-update of tile t-1 co-executes with
// the 6 XDL WMMAs of tile t.
// ---------------------------------------------------------------------------
__global__ __launch_bounds__(256) void k_argmin(const float* __restrict__ x,
                                                const unsigned char* __restrict__ ws_base,
                                                float* __restrict__ idx_f) {
  __shared__ __align__(16) unsigned char lds_all[LDS_TOTAL];
  uint4* lds_cb   = (uint4*)lds_all;
  float* lds_norm = (float*)(lds_all + CB_PACKED_BYTES);

  // ---- one TDM op copies packed codebook + norms (266240 B) into LDS ----
  if (threadIdx.x < 32) {
    unsigned lds_base = (unsigned)(size_t)(void*)lds_all;     // LDS byte offset
    unsigned long long g = (unsigned long long)(size_t)ws_base;
    v4u g0 = { 1u,                                            // count=1 (valid)
               lds_base,                                      // lds_addr
               (unsigned)(g & 0xFFFFFFFFu),                   // global_addr lo
               (unsigned)((g >> 32) & 0x1FFFFFFu) | (2u << 30) }; // hi | type=2
    v8i g1 = { (int)(3u << 16),        // data_size=3 (8B), no mask/iter/pad
               (int)((TDM_ELEMS & 0xFFFF) << 16),   // tensor_dim0[15:0]
               (int)((TDM_ELEMS >> 16) | (1u << 16)),// t_dim0 hi | tensor_dim1=1
               (int)((TDM_ELEMS & 0xFFFF) << 16),   // tile_dim0 = 33280
               1,                                   // tile_dim1 = 1
               (int)TDM_ELEMS,                      // tensor_dim0_stride lo
               (int)((TDM_ELEMS & 0xFFFF) << 16),   // stride hi=0 | dim1_stride
               0 };
    v4i gz4 = { 0, 0, 0, 0 };
    v8i gz8 = { 0, 0, 0, 0, 0, 0, 0, 0 };
    __builtin_amdgcn_tensor_load_to_lds(g0, g1, gz4, gz4, gz8, 0);
    __builtin_amdgcn_s_wait_tensorcnt(0);
  }
  __syncthreads();

  const int tid  = threadIdx.x;
  const int lane = tid & 31;
  const int wave = tid >> 5;
  const int m    = lane & 15;            // A row / C code column
  const int koff = (lane >> 4) << 3;     // A-layout K offset for upper half

  for (int rt = 0; rt < 4; ++rt) {
    const int tok0 = blockIdx.x * 512 + (wave * 4 + rt) * 16;
    const float* row = x + (size_t)(tok0 + m) * DIM;

    // ---- pack A tile (16 tokens x 64 dims) into hi/lo v16bf per K-chunk ----
    v16bf ahi[2], alo[2];
    #pragma unroll
    for (int c = 0; c < 2; ++c) {
      const float4* p1 = (const float4*)(row + c * 32 + 2 * koff);
      const float4* p2 = (const float4*)(row + c * 32 + 8 + 2 * koff);
      float4 f0 = p1[0], f1 = p1[1], f2 = p2[0], f3 = p2[1];
      float va[16] = {f0.x, f0.y, f0.z, f0.w, f1.x, f1.y, f1.z, f1.w,
                      f2.x, f2.y, f2.z, f2.w, f3.x, f3.y, f3.z, f3.w};
      #pragma unroll
      for (int e = 0; e < 16; ++e) {
        __bf16 h = (__bf16)va[e];
        ahi[c][e] = h;
        alo[c][e] = (__bf16)(va[e] - (float)h);
      }
    }

    float minv[8];
    int   mini[8];
    #pragma unroll
    for (int q = 0; q < 8; ++q) { minv[q] = 3.4e38f; mini[q] = 0; }

    // 6 WMMAs for one 16x16 code tile
    auto tile_acc = [&](int t) -> v8f {
      v8f a = {0.f, 0.f, 0.f, 0.f, 0.f, 0.f, 0.f, 0.f};
      #pragma unroll
      for (int c = 0; c < 2; ++c) {
        const int base = (t * 2 + c) * 128;           // uint4 units
        union { uint4 u[2]; v16bf v; } bh, bl;
        bh.u[0] = lds_cb[base +       lane];
        bh.u[1] = lds_cb[base + 32  + lane];
        bl.u[0] = lds_cb[base + 64  + lane];
        bl.u[1] = lds_cb[base + 96  + lane];
        a = __builtin_amdgcn_wmma_f32_16x16x32_bf16(false, ahi[c], false, bh.v,
                                                    (short)0, a, false, false);
        a = __builtin_amdgcn_wmma_f32_16x16x32_bf16(false, ahi[c], false, bl.v,
                                                    (short)0, a, false, false);
        a = __builtin_amdgcn_wmma_f32_16x16x32_bf16(false, alo[c], false, bh.v,
                                                    (short)0, a, false, false);
      }
      return a;
    };
    // argmin update for a finished tile (runs while next tile's WMMAs drain)
    auto upd = [&](const v8f& a, int t) {
      const int code = t * 16 + m;
      const float cn = lds_norm[code];
      #pragma unroll
      for (int q = 0; q < 8; ++q) {
        float sc = __builtin_fmaf(-2.0f, a[q], cn); // |e|^2 - 2 x.e
        if (sc < minv[q]) { minv[q] = sc; mini[q] = code; }
      }
    };

    // ---- software-pipelined sweep over all 64 code column tiles ----
    v8f accp = tile_acc(0);
    for (int t = 1; t < N_KTILES; ++t) {
      v8f accn = tile_acc(t);
      upd(accp, t - 1);
      accp = accn;
    }
    upd(accp, N_KTILES - 1);

    // ---- reduce across the 16 lanes of each half (rows 0-7 / 8-15) ----
    #pragma unroll
    for (int q = 0; q < 8; ++q) {
      float v = minv[q];
      int  ix = mini[q];
      #pragma unroll
      for (int off = 1; off < 16; off <<= 1) {
        float ov = __shfl_xor(v, off, 32);
        int   oi = __shfl_xor(ix, off, 32);
        if (ov < v || (ov == v && oi < ix)) { v = ov; ix = oi; }
      }
      if (m == 0) {
        int rowm = q + ((lane >> 4) << 3);   // C layout: row = q + 8*(lane>=16)
        idx_f[tok0 + rowm] = (float)ix;
      }
    }
  }
}

// ---------------------------------------------------------------------------
// k_gather: quantized_st forward value == gathered codebook row; deterministic
// per-block MSE partial sums; exact integer histogram atomics.
// ---------------------------------------------------------------------------
__global__ __launch_bounds__(256) void k_gather(const float* __restrict__ x,
                                                const float* __restrict__ cb,
                                                const float* __restrict__ idx_f,
                                                float* __restrict__ outq,
                                                int* __restrict__ hist,
                                                float* __restrict__ part) {
  __shared__ float red[256];
  int tid   = threadIdx.x;
  int token = blockIdx.x * 256 + tid;
  int idx   = (int)idx_f[token];
  const float4* q4 = (const float4*)(cb + (size_t)idx * DIM);
  const float4* x4 = (const float4*)(x + (size_t)token * DIM);
  float4*       o4 = (float4*)(outq + (size_t)token * DIM);
  float s = 0.f;
  #pragma unroll
  for (int d = 0; d < 16; ++d) {
    float4 q = q4[d], xi = x4[d];
    o4[d] = q;
    float dx = q.x - xi.x, dy = q.y - xi.y, dz = q.z - xi.z, dw = q.w - xi.w;
    s += dx * dx + dy * dy + dz * dz + dw * dw;
  }
  atomicAdd(&hist[idx], 1);
  red[tid] = s;
  __syncthreads();
  for (int o = 128; o > 0; o >>= 1) {
    if (tid < o) red[tid] += red[tid + o];
    __syncthreads();
  }
  if (tid == 0) part[blockIdx.x] = red[0];
}

// ---------------------------------------------------------------------------
// k_final: loss = 1.25 * mse ; perplexity = exp(-sum p*log(p+1e-10))
// ---------------------------------------------------------------------------
__global__ __launch_bounds__(1024) void k_final(const int* __restrict__ hist,
                                                const float* __restrict__ part,
                                                float* __restrict__ out) {
  __shared__ float red[1024];
  int tid = threadIdx.x;
  float p = (float)hist[tid] * (1.0f / (float)N_TOKENS);
  red[tid] = p * logf(p + 1e-10f);
  __syncthreads();
  for (int o = 512; o > 0; o >>= 1) {
    if (tid < o) red[tid] += red[tid + o];
    __syncthreads();
  }
  if (tid == 0) out[1 + (size_t)N_TOKENS * DIM] = expf(-red[0]);
  __syncthreads();
  red[tid] = (tid < NBLK_K2) ? part[tid] : 0.f;
  __syncthreads();
  for (int o = 512; o > 0; o >>= 1) {
    if (tid < o) red[tid] += red[tid + o];
    __syncthreads();
  }
  if (tid == 0) {
    float mse = red[0] / ((float)N_TOKENS * (float)DIM);
    out[0] = 1.25f * mse;   // mse + COMMITMENT_COST * mse
  }
}

// ---------------------------------------------------------------------------
extern "C" void kernel_launch(void* const* d_in, const int* in_sizes, int n_in,
                              void* d_out, int out_size, void* d_ws, size_t ws_size,
                              hipStream_t stream) {
  const float* x  = (const float*)d_in[0];   // [131072, 64]
  const float* cb = (const float*)d_in[1];   // [1024, 64]
  float* out = (float*)d_out;
  char*  ws  = (char*)d_ws;
  __bf16* cbp  = (__bf16*)(ws + WS_CB_PACK);
  float*  cbn  = (float*)(ws + WS_CB_NORM);
  int*    hist = (int*)(ws + WS_HIST);
  float*  part = (float*)(ws + WS_PART);
  // d_out layout: [0]=loss, [1 .. N*D]=quantized_st, [1+N*D]=perplexity,
  //               [2+N*D .. 2+N*D+N-1]=indices (as float)
  float* outq = out + 1;
  float* idxf = out + 2 + (size_t)N_TOKENS * DIM;

  k_pack  <<<512, 256, 0, stream>>>(cb, cbp);
  k_norm  <<<4,   256, 0, stream>>>(cb, cbn, hist, part);
  k_argmin<<<256, 256, 0, stream>>>(x, (const unsigned char*)ws, idxf);
  k_gather<<<512, 256, 0, stream>>>(x, cb, idxf, outq, hist, part);
  k_final <<<1,  1024, 0, stream>>>(hist, part, out);
}
